// SimpleLSTM_2147483648036
// MI455X (gfx1250) — matrix operand order repaired and checked by reading, not compiled, not run on hardware
//
#include <hip/hip_runtime.h>
#include <hip/hip_bf16.h>

// ---------------------------------------------------------------------------
// SimpleLSTM on MI455X (gfx1250): persistent per-workgroup recurrence using
// v_wmma_f32_16x16x32_f16 for the per-step 16x1024x256 GEMM (per block).
// Batch rows are independent -> each 256-thread block owns 16 rows for all
// 784 timesteps. Weights pre-converted to f16 and pre-swizzled into WMMA
// B-fragment layout in d_ws. Activations use hardware v_tanh_f32. gx lives
// permanently in VGPRs and is folded into the first WMMA's C operand.
// ---------------------------------------------------------------------------

typedef __attribute__((ext_vector_type(16))) _Float16 v16h;
typedef __attribute__((ext_vector_type(8)))  _Float16 v8h;
typedef __attribute__((ext_vector_type(8)))  float    v8f;
typedef __attribute__((ext_vector_type(4)))  float    v4f;

#define LSTM_B     1024
#define LSTM_D     784      // timesteps = 28*28
#define LSTM_DP    800      // D padded to multiple of 32
#define LSTM_H     256
#define LSTM_G     1024     // 4*H gates
#define ROWS_PB    16       // batch rows per block
#define KC_HH      8        // 256 / 32
#define KC_IH      25       // 800 / 32

// LDS layout: gates stored transposed [N=1024][M=16 + pad 4] so WMMA
// accumulators write as b128 and the elementwise phase reads b128.
#define GPAD       20                          // floats per gate column
#define SZ_GATES   (LSTM_G * GPAD * 4)         // 81920 B
#define OFF_H      SZ_GATES
#define HSTRIDE    264                         // halves per h row (256+8)
#define OFF_V      (OFF_H + ROWS_PB * HSTRIDE * 2)      // 90368
#define VSTRIDE    808                         // halves per v row (800+8)
#define DYN_LDS    (OFF_V + ROWS_PB * VSTRIDE * 2)      // 116224 B

union AV  { v16h v; v8h h[2]; };
union ACC { v8f  v; v4f q[2]; };

__device__ __forceinline__ float fast_tanh(float x) {
#if __has_builtin(__builtin_amdgcn_tanhf)
    return __builtin_amdgcn_tanhf(x);      // v_tanh_f32 (TRANS, co-executes w/ WMMA)
#else
    return tanhf(x);
#endif
}
__device__ __forceinline__ float sigm(float x) {
#if __has_builtin(__builtin_amdgcn_tanhf)
    return 0.5f + 0.5f * __builtin_amdgcn_tanhf(0.5f * x);
#else
    return 1.0f / (1.0f + __expf(-x));
#endif
}

// --- pre-swizzle w_ih [1024 x 784] f32 -> f16 B-fragments, K zero-padded ----
// dst index = ((tn*25 + kc)*32 + lane)*16 + q ; tn = wave*8 + j
// lane<16:  N = tn*16+lane,     K = kc*32 + q
// lane>=16: N = tn*16+lane-16,  K = kc*32 + 16 + q
__global__ void swz_wih_kernel(const float* __restrict__ w, _Float16* __restrict__ dst) {
    int idx = blockIdx.x * 256 + threadIdx.x;
    if (idx >= 64 * KC_IH * 512) return;
    int q    = idx & 15;
    int lane = (idx >> 4) & 31;
    int t    = idx >> 9;          // tn*25 + kc
    int kc   = t % KC_IH;
    int tn   = t / KC_IH;
    int n = tn * 16 + (lane & 15);
    int k = kc * 32 + ((lane >> 4) << 4) + q;
    float v = (k < LSTM_D) ? w[n * LSTM_D + k] : 0.0f;
    dst[idx] = (_Float16)v;
}

// --- pre-swizzle w_hh [1024 x 256] f32 -> f16 B-fragments ------------------
__global__ void swz_whh_kernel(const float* __restrict__ w, _Float16* __restrict__ dst) {
    int idx = blockIdx.x * 256 + threadIdx.x;
    if (idx >= 64 * KC_HH * 512) return;
    int q    = idx & 15;
    int lane = (idx >> 4) & 31;
    int t    = idx >> 9;          // tn*8 + kc
    int kc   = t & 7;
    int tn   = t >> 3;
    int n = tn * 16 + (lane & 15);
    int k = kc * 32 + ((lane >> 4) << 4) + q;
    dst[idx] = (_Float16)w[n * LSTM_H + k];
}

// --- main persistent LSTM kernel -------------------------------------------
__global__ void __launch_bounds__(256) lstm_main_kernel(
    const float* __restrict__ x,
    const float* __restrict__ b_ih,
    const float* __restrict__ b_hh,
    const float* __restrict__ w_out,
    const float* __restrict__ b_out,
    const _Float16* __restrict__ wsWih,
    const _Float16* __restrict__ wsWhh,
    float* __restrict__ out)
{
    extern __shared__ __align__(16) char smem[];
    float*    sh_gates = (float*)smem;                 // [1024][GPAD]
    _Float16* sh_h     = (_Float16*)(smem + OFF_H);    // [16][HSTRIDE]
    _Float16* sh_v     = (_Float16*)(smem + OFF_V);    // [16][VSTRIDE] (phase 1)
    float*    sh_pp    = (float*)(smem + OFF_V);       // [16][256] (reused)
    float*    sh_p     = (float*)(smem + OFF_V + ROWS_PB * LSTM_H * 4); // [256]

    const int tid  = threadIdx.x;
    const int lane = tid & 31;
    const int wave = tid >> 5;
    const int arow = lane & 15;       // A-fragment row (M)
    const int asel = lane >> 4;       // A-fragment K-half select
    const int r0   = blockIdx.x * ROWS_PB;

    // per-(wave,lane) weight fragment base pointers; element (j,kc) lives at
    // byte offset (j*8+kc)*1024 (+16) -> folds into b128 immediate offsets.
    const v8h* __restrict__ bbHH = (const v8h*)(wsWhh + (size_t)wave * 32768 + (size_t)lane * 16);
    const v8h* __restrict__ bbIH = (const v8h*)(wsWih + (size_t)wave * 102400 + (size_t)lane * 16);

    // ---- stage flattened input rows (f32 -> f16, zero-padded K) into LDS ----
    for (int idx = tid; idx < ROWS_PB * LSTM_DP; idx += 256) {
        int m = idx / LSTM_DP;
        int k = idx - m * LSTM_DP;
        float v = (k < LSTM_D) ? x[(size_t)(r0 + m) * LSTM_D + k] : 0.0f;
        sh_v[m * VSTRIDE + k] = (_Float16)v;
    }
    for (int idx = tid; idx < ROWS_PB * HSTRIDE; idx += 256)
        sh_h[idx] = (_Float16)0.0f;
    __syncthreads();

    // ---- phase 1: gx = v @ w_ih^T + (b_ih + b_hh), kept in VGPRs ----------
    v8f gx[8];
#pragma unroll
    for (int j = 0; j < 8; ++j) gx[j] = (v8f){};

    for (int kc = 0; kc < KC_IH; ++kc) {
        const int K0 = kc * 32;
        AV a;
        a.h[0] = *(const v8h*)&sh_v[arow * VSTRIDE + K0 + asel * 16];
        a.h[1] = *(const v8h*)&sh_v[arow * VSTRIDE + K0 + asel * 16 + 8];
        AV b[8];
#pragma unroll
        for (int j = 0; j < 8; ++j) {
            b[j].h[0] = bbIH[(j * KC_IH + kc) * 64];
            b[j].h[1] = bbIH[(j * KC_IH + kc) * 64 + 1];
        }
#pragma unroll
        for (int j = 0; j < 8; ++j)
            gx[j] = __builtin_amdgcn_wmma_f32_16x16x32_f16(
                        false, a.v, false, b[j].v, (short)0, gx[j], false, false);
    }
#pragma unroll
    for (int j = 0; j < 8; ++j) {
        const int n = (wave * 8 + j) * 16 + (lane & 15);
        const float bias = b_ih[n] + b_hh[n];
        gx[j] += bias;   // splat: bias depends on N only
    }

    // per-thread recurrent state: thread t owns hidden column t, rows 0..15
    float cst[ROWS_PB];
#pragma unroll
    for (int m = 0; m < ROWS_PB; ++m) cst[m] = 0.0f;
    const float wout_t = w_out[tid];
    const float b0     = b_out[0];

    __syncthreads();

    // ---- recurrent loop over 784 timesteps --------------------------------
    for (int step = 0; step < LSTM_D; ++step) {
        ACC acc[8];

        // kc = 0: consume gx directly as WMMA C operand (no copy into acc)
        {
            AV a;
            a.h[0] = *(const v8h*)&sh_h[arow * HSTRIDE + asel * 16];
            a.h[1] = *(const v8h*)&sh_h[arow * HSTRIDE + asel * 16 + 8];
            AV b[8];
#pragma unroll
            for (int j = 0; j < 8; ++j) {
                b[j].h[0] = bbHH[(j * KC_HH) * 64];
                b[j].h[1] = bbHH[(j * KC_HH) * 64 + 1];
            }
#pragma unroll
            for (int j = 0; j < 8; ++j)
                acc[j].v = __builtin_amdgcn_wmma_f32_16x16x32_f16(
                               false, a.v, false, b[j].v, (short)0, gx[j], false, false);
        }

#pragma unroll
        for (int kc = 1; kc < KC_HH; ++kc) {
            const int K0 = kc * 32;
            AV a;
            a.h[0] = *(const v8h*)&sh_h[arow * HSTRIDE + K0 + asel * 16];
            a.h[1] = *(const v8h*)&sh_h[arow * HSTRIDE + K0 + asel * 16 + 8];
            if (kc + 1 < KC_HH)
                __builtin_prefetch((const void*)&bbHH[(kc + 1) * 64], 0, 1);
            AV b[8];
#pragma unroll
            for (int j = 0; j < 8; ++j) {
                b[j].h[0] = bbHH[(j * KC_HH + kc) * 64];
                b[j].h[1] = bbHH[(j * KC_HH + kc) * 64 + 1];
            }
#pragma unroll
            for (int j = 0; j < 8; ++j)
                acc[j].v = __builtin_amdgcn_wmma_f32_16x16x32_f16(
                               false, a.v, false, b[j].v, (short)0, acc[j].v, false, false);
        }

        // scatter accumulators to LDS transposed gates buffer as b128:
        // C layout: reg r -> M = r + 8*asel, lane -> N; [N][M] w/ stride GPAD.
#pragma unroll
        for (int j = 0; j < 8; ++j) {
            const int n = (wave * 8 + j) * 16 + (lane & 15);
            *(v4f*)&sh_gates[n * GPAD + asel * 8]     = acc[j].q[0];
            *(v4f*)&sh_gates[n * GPAD + asel * 8 + 4] = acc[j].q[1];
        }
        __syncthreads();

        // elementwise LSTM update: thread t handles hidden column t, rows 0..15
#pragma unroll
        for (int mg = 0; mg < 4; ++mg) {
            const v4f gi4 = *(const v4f*)&sh_gates[(0   + tid) * GPAD + mg * 4];
            const v4f gf4 = *(const v4f*)&sh_gates[(256 + tid) * GPAD + mg * 4];
            const v4f gg4 = *(const v4f*)&sh_gates[(512 + tid) * GPAD + mg * 4];
            const v4f go4 = *(const v4f*)&sh_gates[(768 + tid) * GPAD + mg * 4];
#pragma unroll
            for (int e = 0; e < 4; ++e) {
                const int m = mg * 4 + e;
                const float iv = sigm(gi4[e]);
                const float fv = sigm(gf4[e]);
                const float gv = fast_tanh(gg4[e]);
                const float ov = sigm(go4[e]);
                cst[m] = fv * cst[m] + iv * gv;
                const float hv = ov * fast_tanh(cst[m]);
                sh_h[m * HSTRIDE + tid] = (_Float16)hv;
                sh_pp[m * LSTM_H + tid] = hv * wout_t;   // partial for out proj
            }
        }
        __syncthreads();

        // output projection reduction, stage 1: 256 -> 16 partials per row
        {
            const int m2 = tid >> 4;
            const int g2 = tid & 15;
            float s = 0.0f;
#pragma unroll
            for (int q = 0; q < 16; ++q) s += sh_pp[m2 * LSTM_H + g2 * 16 + q];
            sh_p[tid] = s;
        }
        __syncthreads();

        // stage 2: 16 -> 1, store out[b, step]
        if (tid < ROWS_PB) {
            float s = 0.0f;
#pragma unroll
            for (int q = 0; q < 16; ++q) s += sh_p[tid * 16 + q];
            out[(size_t)(r0 + tid) * LSTM_D + step] = s + b0;
        }
        // no barrier needed: next matmul only reads sh_h / writes sh_gates,
        // both fenced relative to this step's readers by the syncs above.
    }
}

extern "C" void kernel_launch(void* const* d_in, const int* in_sizes, int n_in,
                              void* d_out, int out_size, void* d_ws, size_t ws_size,
                              hipStream_t stream) {
    const float* x     = (const float*)d_in[0];   // [1024,28,28]
    const float* w_ih  = (const float*)d_in[1];   // [1024,784]
    const float* w_hh  = (const float*)d_in[2];   // [1024,256]
    const float* b_ih  = (const float*)d_in[3];   // [1024]
    const float* b_hh  = (const float*)d_in[4];   // [1024]
    const float* w_out = (const float*)d_in[5];   // [1,256]
    const float* b_out = (const float*)d_in[6];   // [1]
    float* out = (float*)d_out;                   // [1024,784] flat

    // workspace: f16 swizzled weights (w_ih: 1.6MB, w_hh: 512KB)
    _Float16* wsWih = (_Float16*)d_ws;
    _Float16* wsWhh = wsWih + (size_t)64 * KC_IH * 512;   // 819200 halves

    swz_wih_kernel<<<(64 * KC_IH * 512 + 255) / 256, 256, 0, stream>>>(w_ih, wsWih);
    swz_whh_kernel<<<(64 * KC_HH * 512 + 255) / 256, 256, 0, stream>>>(w_hh, wsWhh);

    (void)hipFuncSetAttribute((const void*)lstm_main_kernel,
                              hipFuncAttributeMaxDynamicSharedMemorySize, DYN_LDS);

    lstm_main_kernel<<<LSTM_B / ROWS_PB, 256, DYN_LDS, stream>>>(
        x, b_ih, b_hh, w_out, b_out, wsWih, wsWhh, out);
}
